// MultiHeadedAttention_55559696941527
// MI455X (gfx1250) — matrix operand order repaired and checked by reading, not compiled, not run on hardware
//
#include <hip/hip_runtime.h>
#include <hip/hip_bf16.h>

#define T_SEQ 4096
#define EMB   2048
#define NH    16
#define HD    128

typedef __attribute__((ext_vector_type(16))) __bf16 v16bf;
typedef __attribute__((ext_vector_type(8)))  float  v8f;
typedef int v4i __attribute__((vector_size(16)));

struct alignas(16) U4 { unsigned x, y, z, w; };
struct F4 { float x, y, z, w; };
struct S4 { unsigned short a, b, c, e; };
union AFrag { v16bf v; U4 q[2]; };

__device__ __forceinline__ unsigned short f2bf_raw(float f) {
  unsigned u = __builtin_bit_cast(unsigned, f);
  unsigned r = u + 0x7fffu + ((u >> 16) & 1u);      // round-to-nearest-even
  return (unsigned short)(r >> 16);
}

__device__ __forceinline__ v8f wmma_bf16(const v16bf& a, const v16bf& b, const v8f& c) {
  return __builtin_amdgcn_wmma_f32_16x16x32_bf16(false, a, false, b, (short)0, c, false, false);
}

// 16-byte async copy global -> LDS (per-lane addresses), tracked by ASYNCcnt.
__device__ __forceinline__ void async_cp16(const unsigned short* g, unsigned short* l) {
#if __has_builtin(__builtin_amdgcn_global_load_async_to_lds_b128)
  __builtin_amdgcn_global_load_async_to_lds_b128(
      (__attribute__((address_space(1))) v4i*)(unsigned short*)g,
      (__attribute__((address_space(3))) v4i*)l, 0, 0);
#else
  unsigned lo = (unsigned)(size_t)(__attribute__((address_space(3))) unsigned short*)l;
  asm volatile("global_load_async_to_lds_b128 %0, %1, off" :: "v"(lo), "v"(g) : "memory");
#endif
}

__device__ __forceinline__ void wait_async0() {
#if __has_builtin(__builtin_amdgcn_s_wait_asynccnt)
  __builtin_amdgcn_s_wait_asynccnt(0);
#else
  asm volatile("s_wait_asynccnt 0" ::: "memory");
#endif
}

// ---------------------------------------------------------------------------
// GEMM: C[4096 x 2048] = A[4096 x 2048] * B[2048 x 2048]^T   (x @ W^T)
// Software-pipelined: next k-tile global loads overlap WMMA on current tile.
// OUT_MODE 0: bf16 row-major | 1: bf16 per-head transposed | 2: f32 + residual
// ---------------------------------------------------------------------------
template <int OUT_MODE, typename AT>
__global__ __launch_bounds__(256) void gemm_kernel(
    const AT* __restrict__ A, const float* __restrict__ B,
    const float* __restrict__ resid, void* __restrict__ Cout) {
  constexpr int LDA = 40;                       // 80B row stride (16B multiple)
  __shared__ unsigned short As[128 * LDA];
  __shared__ unsigned short Bs[128 * LDA];

  const int tid  = threadIdx.x;
  const int wid  = tid >> 5;
  const int lane = tid & 31;
  const int r16  = lane & 15, hi = lane >> 4;
  const int mBase = blockIdx.x * 128;
  const int nBase = blockIdx.y * 128;
  const int wm = (wid >> 1) * 32;
  const int wn = (wid & 1) * 64;

  const int sRow = tid >> 3;                    // 0..31
  const int sCol = (tid & 7) * 4;               // 0..28

  v8f acc[2][4] = {};

  F4 pb[4];
  F4 paf[4];   // A prefetch when AT=float
  S4 pab[4];   // A prefetch when AT=bf16

  auto loadTile = [&](int k0) {
#pragma unroll
    for (int rr = 0; rr < 4; rr++) {
      const int row = sRow + rr * 32;
      if constexpr (sizeof(AT) == 4)
        paf[rr] = *reinterpret_cast<const F4*>(A + (size_t)(mBase + row) * EMB + k0 + sCol);
      else
        pab[rr] = *reinterpret_cast<const S4*>(A + (size_t)(mBase + row) * EMB + k0 + sCol);
      pb[rr] = *reinterpret_cast<const F4*>(B + (size_t)(nBase + row) * EMB + k0 + sCol);
    }
  };

  loadTile(0);

  for (int k0 = 0; k0 < EMB; k0 += 32) {
    __syncthreads();                            // prior fragment reads done
#pragma unroll
    for (int rr = 0; rr < 4; rr++) {
      const int row = sRow + rr * 32;
      unsigned short* da = &As[row * LDA + sCol];
      if constexpr (sizeof(AT) == 4) {
        da[0] = f2bf_raw(paf[rr].x); da[1] = f2bf_raw(paf[rr].y);
        da[2] = f2bf_raw(paf[rr].z); da[3] = f2bf_raw(paf[rr].w);
      } else {
        *reinterpret_cast<S4*>(da) = pab[rr];
      }
      unsigned short* db = &Bs[row * LDA + sCol];
      db[0] = f2bf_raw(pb[rr].x); db[1] = f2bf_raw(pb[rr].y);
      db[2] = f2bf_raw(pb[rr].z); db[3] = f2bf_raw(pb[rr].w);
    }
    __syncthreads();

    if (k0 + 32 < EMB) loadTile(k0 + 32);       // overlap with WMMAs below

    AFrag af[2];
#pragma unroll
    for (int i = 0; i < 2; i++) {
      const unsigned short* base = &As[(wm + i * 16 + r16) * LDA];
      af[i].q[0] = *reinterpret_cast<const U4*>(base + hi * 8);
      af[i].q[1] = *reinterpret_cast<const U4*>(base + 16 + hi * 8);
    }
    AFrag bfr[4];
#pragma unroll
    for (int j = 0; j < 4; j++) {
      const unsigned short* base = &Bs[(wn + j * 16 + r16) * LDA] + hi * 8;
      bfr[j].q[0] = *reinterpret_cast<const U4*>(base);
      bfr[j].q[1] = *reinterpret_cast<const U4*>(base + 16);
    }
#pragma unroll
    for (int i = 0; i < 2; i++)
#pragma unroll
      for (int j = 0; j < 4; j++)
        acc[i][j] = wmma_bf16(af[i].v, bfr[j].v, acc[i][j]);
  }

#pragma unroll
  for (int i = 0; i < 2; i++)
#pragma unroll
    for (int j = 0; j < 4; j++)
#pragma unroll
      for (int e = 0; e < 8; e++) {
        const int m = mBase + wm + i * 16 + e + 8 * hi;
        const int n = nBase + wn + j * 16 + r16;
        const float val = acc[i][j][e];
        if constexpr (OUT_MODE == 0) {
          ((unsigned short*)Cout)[(size_t)m * EMB + n] = f2bf_raw(val);
        } else if constexpr (OUT_MODE == 1) {
          const int h = n >> 7, d = n & 127;
          ((unsigned short*)Cout)[(size_t)(h * HD + d) * T_SEQ + m] = f2bf_raw(val);
        } else {
          ((float*)Cout)[(size_t)m * EMB + n] = val + resid[(size_t)m * EMB + n];
        }
      }
}

// ---------------------------------------------------------------------------
// Causal flash attention. One wave = 16 query rows of one head.
// K/V tiles double-buffered in LDS via async global->LDS copies (ASYNCcnt).
// Dynamic LDS layout (bf16 elems):
//   Ks[2][64 *136]  @ 0          (row stride 272B)
//   Vs[2][128* 72]  @ 17408      (row stride 144B)
//   Ps[8][16 * 72]  @ 35840      (per-wave P spill)
// total 45056 elems = 90112 B
// ---------------------------------------------------------------------------
__global__ __launch_bounds__(256) void attn_kernel(
    const unsigned short* __restrict__ Qb, const unsigned short* __restrict__ Kb,
    const unsigned short* __restrict__ Vt, unsigned short* __restrict__ Yb) {
  constexpr int KSTR = 136, VSTR = 72, PSTR = 72;
  constexpr int KS_BUF = 64 * KSTR;             // 8704
  constexpr int VS_BASE = 2 * KS_BUF;           // 17408
  constexpr int VS_BUF = 128 * VSTR;            // 9216
  constexpr int PS_BASE = VS_BASE + 2 * VS_BUF; // 35840
  extern __shared__ __align__(16) char smem_raw[];
  unsigned short* smem = (unsigned short*)smem_raw;

  const int tid  = threadIdx.x;
  const int wid  = tid >> 5;
  const int lane = tid & 31;
  const int r16  = lane & 15, hi = lane >> 4;
  const int h  = blockIdx.y;
  const int q0 = blockIdx.x * 128 + wid * 16;
  const float scale = 0.08838834764831845f;     // 1/sqrt(128)
  unsigned short* myP = smem + PS_BASE + wid * 16 * PSTR;

  // issue async stage of one 64-key block (K: 64x128, V: 128x64) into buf
  auto stage = [&](int buf, int s0) {
    unsigned short* ks = smem + buf * KS_BUF;
    unsigned short* vs = smem + VS_BASE + buf * VS_BUF;
#pragma unroll
    for (int t = 0; t < 4; t++) {
      const int ck = tid + t * 256;             // 0..1023
      const int kr = ck >> 4, kc = (ck & 15) * 8;
      async_cp16(Kb + (size_t)(s0 + kr) * EMB + h * HD + kc, ks + kr * KSTR + kc);
      const int vr = ck >> 3, vc = (ck & 7) * 8;
      async_cp16(Vt + (size_t)(h * HD + vr) * T_SEQ + s0 + vc, vs + vr * VSTR + vc);
    }
  };

  // Q fragments (16 x 128) resident for the whole pass
  AFrag qf[4];
  {
    const unsigned short* qr = Qb + (size_t)(q0 + r16) * EMB + h * HD;
#pragma unroll
    for (int c = 0; c < 4; c++) {
      qf[c].q[0] = *reinterpret_cast<const U4*>(qr + c * 32 + hi * 8);
      qf[c].q[1] = *reinterpret_cast<const U4*>(qr + c * 32 + 16 + hi * 8);
    }
  }

  v8f yacc[8] = {};
  float mi[8], li[8];
#pragma unroll
  for (int e = 0; e < 8; e++) { mi[e] = -1e30f; li[e] = 0.0f; }

  const int qmax = q0 + 15;
  const int nblk = blockIdx.x * 2 + 2;          // block-uniform trip count

  stage(0, 0);
  for (int blk = 0; blk < nblk; blk++) {
    const int s0  = blk * 64;
    const int cur = blk & 1;
    wait_async0();                              // my async copies landed
    __syncthreads();                            // everyone's landed
    if (blk + 1 < nblk) stage(cur ^ 1, (blk + 1) * 64);

    if (s0 <= qmax) {                           // wave-uniform causal skip
      const unsigned short* ks = smem + cur * KS_BUF;
      const unsigned short* vs = smem + VS_BASE + cur * VS_BUF;

      // ---- S = Q K^T (16 x 64) ----
      v8f sacc[4] = {};
#pragma unroll
      for (int j = 0; j < 4; j++)
#pragma unroll
        for (int c = 0; c < 4; c++) {
          AFrag kf;
          const unsigned short* kp = ks + (j * 16 + r16) * KSTR + c * 32 + hi * 8;
          kf.q[0] = *reinterpret_cast<const U4*>(kp);
          kf.q[1] = *reinterpret_cast<const U4*>(kp + 16);
          sacc[j] = wmma_bf16(qf[c].v, kf.v, sacc[j]);
        }

      // ---- scale + causal mask + block row-max ----
      float bm[8];
#pragma unroll
      for (int e = 0; e < 8; e++) bm[e] = -1e30f;
#pragma unroll
      for (int j = 0; j < 4; j++)
#pragma unroll
        for (int e = 0; e < 8; e++) {
          const int qrow = q0 + e + 8 * hi;
          const int srow = s0 + j * 16 + r16;
          float sv = sacc[j][e] * scale;
          sv = (srow <= qrow) ? sv : -1e30f;
          sacc[j][e] = sv;
          bm[e] = fmaxf(bm[e], sv);
        }
#pragma unroll
      for (int off = 1; off < 16; off <<= 1)
#pragma unroll
        for (int e = 0; e < 8; e++) bm[e] = fmaxf(bm[e], __shfl_xor(bm[e], off, 32));

      float alpha[8];
#pragma unroll
      for (int e = 0; e < 8; e++) {
        const float mn = fmaxf(mi[e], bm[e]);
        alpha[e] = __expf(mi[e] - mn);
        mi[e] = mn;
      }

      // ---- P = exp(S - m), row sums, spill P to LDS (C- -> A-layout) ----
      float bs[8] = {};
#pragma unroll
      for (int j = 0; j < 4; j++)
#pragma unroll
        for (int e = 0; e < 8; e++) {
          const float p = __expf(sacc[j][e] - mi[e]);
          bs[e] += p;
          myP[(e + 8 * hi) * PSTR + j * 16 + r16] = f2bf_raw(p);
        }
#pragma unroll
      for (int off = 1; off < 16; off <<= 1)
#pragma unroll
        for (int e = 0; e < 8; e++) bs[e] += __shfl_xor(bs[e], off, 32);
#pragma unroll
      for (int e = 0; e < 8; e++) li[e] = li[e] * alpha[e] + bs[e];

#pragma unroll
      for (int t = 0; t < 8; t++)
#pragma unroll
        for (int e = 0; e < 8; e++) yacc[t][e] *= alpha[e];

      asm volatile("s_wait_dscnt 0" ::: "memory");

      // ---- y += P V ----
      AFrag pf[2];
#pragma unroll
      for (int c = 0; c < 2; c++) {
        const unsigned short* pr = myP + r16 * PSTR + c * 32;
        pf[c].q[0] = *reinterpret_cast<const U4*>(pr + hi * 8);
        pf[c].q[1] = *reinterpret_cast<const U4*>(pr + 16 + hi * 8);
      }
#pragma unroll
      for (int t = 0; t < 8; t++)
#pragma unroll
        for (int c = 0; c < 2; c++) {
          AFrag vf;
          const unsigned short* vp = vs + (t * 16 + r16) * VSTR + c * 32 + hi * 8;
          vf.q[0] = *reinterpret_cast<const U4*>(vp);
          vf.q[1] = *reinterpret_cast<const U4*>(vp + 16);
          yacc[t] = wmma_bf16(pf[c].v, vf.v, yacc[t]);
        }
    }
    __syncthreads();                            // reads done before re-stage
  }

  // ---- normalize and write ----
#pragma unroll
  for (int t = 0; t < 8; t++)
#pragma unroll
    for (int e = 0; e < 8; e++) {
      const int row = q0 + e + 8 * hi;
      const int col = h * HD + t * 16 + r16;
      Yb[(size_t)row * EMB + col] = f2bf_raw(yacc[t][e] / li[e]);
    }
}

// ---------------------------------------------------------------------------
// Per-row LayerNorm (scale only). X already contains y + residual.
// ---------------------------------------------------------------------------
__global__ __launch_bounds__(256) void ln_kernel(
    const float* __restrict__ X, const float* __restrict__ w, float* __restrict__ out) {
  __shared__ float rs[8], rs2[8];
  const int row = blockIdx.x;
  const float* x = X + (size_t)row * EMB;
  float s = 0.0f, s2 = 0.0f;
  for (int i = threadIdx.x; i < EMB; i += 256) {
    const float v = x[i];
    s += v; s2 += v * v;
  }
#pragma unroll
  for (int off = 1; off < 32; off <<= 1) {
    s  += __shfl_xor(s, off, 32);
    s2 += __shfl_xor(s2, off, 32);
  }
  if ((threadIdx.x & 31) == 0) { rs[threadIdx.x >> 5] = s; rs2[threadIdx.x >> 5] = s2; }
  __syncthreads();
  float ts = 0.0f, ts2 = 0.0f;
#pragma unroll
  for (int i = 0; i < 8; i++) { ts += rs[i]; ts2 += rs2[i]; }
  const float mu   = ts * (1.0f / EMB);
  const float var  = ts2 * (1.0f / EMB) - mu * mu;
  const float rstd = rsqrtf(var + 1e-5f);
  for (int i = threadIdx.x; i < EMB; i += 256)
    out[(size_t)row * EMB + i] = (x[i] - mu) * rstd * w[i];
}

extern "C" void kernel_launch(void* const* d_in, const int* in_sizes, int n_in,
                              void* d_out, int out_size, void* d_ws, size_t ws_size,
                              hipStream_t stream) {
  (void)in_sizes; (void)n_in; (void)out_size; (void)ws_size;
  const float* q   = (const float*)d_in[0];
  const float* k   = (const float*)d_in[1];
  const float* v   = (const float*)d_in[2];
  const float* Wq  = (const float*)d_in[3];
  const float* Wk  = (const float*)d_in[4];
  const float* Wv  = (const float*)d_in[5];
  const float* Wo  = (const float*)d_in[6];
  const float* lnw = (const float*)d_in[7];
  // d_in[8] = mask: causal, applied analytically in-kernel.

  char* ws = (char*)d_ws;
  const size_t szBf = (size_t)T_SEQ * EMB * 2;
  unsigned short* Qb = (unsigned short*)(ws);
  unsigned short* Kb = (unsigned short*)(ws + szBf);
  unsigned short* Vt = (unsigned short*)(ws + 2 * szBf);
  unsigned short* Yb = (unsigned short*)(ws + 3 * szBf);
  float*          Pr = (float*)(ws + 4 * szBf);

  const dim3 gg(T_SEQ / 128, EMB / 128);   // 32 x 16 tiles
  const dim3 bb(256);

  gemm_kernel<0, float><<<gg, bb, 0, stream>>>(q, Wq, nullptr, (void*)Qb);
  gemm_kernel<0, float><<<gg, bb, 0, stream>>>(k, Wk, nullptr, (void*)Kb);
  gemm_kernel<1, float><<<gg, bb, 0, stream>>>(v, Wv, nullptr, (void*)Vt);

  const size_t attn_lds = (size_t)(2 * 64 * 136 + 2 * 128 * 72 + 8 * 16 * 72) * 2;
  attn_kernel<<<dim3(T_SEQ / 128, NH), bb, attn_lds, stream>>>(Qb, Kb, Vt, Yb);

  gemm_kernel<2, unsigned short><<<gg, bb, 0, stream>>>(Yb, Wo, q, (void*)Pr);

  ln_kernel<<<dim3(T_SEQ), bb, 0, stream>>>(Pr, lnw, (float*)d_out);
}
